// KACAModule_35974646071390
// MI455X (gfx1250) — matrix operand order repaired and checked by reading, not compiled, not run on hardware
//
#include <hip/hip_runtime.h>
#include <hip/hip_bf16.h>

// ---------------------------------------------------------------------------
// KACA module for MI455X (gfx1250, wave32, WMMA bf16 16x16x32).
//
// Math note: the channel/spatial attention front-end of the reference reduces
// EXACTLY to tf_fused = 2*text, gf_fused = 2*graph (softmax over singleton
// axes == 1.0), so avg_w/avg_b/max_w/max_b/sp_w/sp_b are dead inputs.
// Remaining pipeline: QKV proj -> batch-dim MHA (2048 x (32x32,K=256)) ->
// out proj -> elemwise prod -> lin proj -> gi GEMM (shared) -> fused gh GEMM
// + GRU gates (x2) -> fused dual 7x7 conv + relu + sum.
// GEMMs on v_wmma_f32_16x16x32_bf16; attention staging uses
// global_load_async_to_lds_b128 (ASYNCcnt) for Q/K.
// ---------------------------------------------------------------------------

typedef __attribute__((ext_vector_type(16))) __bf16        v16bf;
typedef __attribute__((ext_vector_type(8)))  float         v8f;
typedef __attribute__((ext_vector_type(4)))  unsigned int  u32x4;
typedef __hip_bfloat16 bf16_t;

#define SEQ 2048
#define BAT 32
#define EMB 256
#define MROWS (BAT * SEQ)   // 65536

// ---- WMMA fragment helpers (ISA 7.12.2 16-bit layouts) ---------------------
// A (16xK tile, row-major source, row stride ld):
//   lane<16 : row=lane,    K chunks {0..7, 16..23}
//   lane>=16: row=lane-16, K chunks {8..15, 24..31}
__device__ __forceinline__ v16bf frag_a(const bf16_t* base, int ld, int lane) {
  const bf16_t* p = base + (size_t)(lane & 15) * ld + ((lane >> 4) << 3);
  union { u32x4 u[2]; v16bf v; } x;
  x.u[0] = *(const u32x4*)(p);
  x.u[1] = *(const u32x4*)(p + 16);
  return x.v;
}
// B (Kx16 tile) sourced from B^T stored row-major (rows = N, cols = K, stride ld):
//   lane<16 : col n=lane,    K chunks {0..7, 16..23}
//   lane>=16: col n=lane-16, K chunks {8..15, 24..31}
__device__ __forceinline__ v16bf frag_b(const bf16_t* base, int ld, int lane) {
  const bf16_t* p = base + (size_t)(lane & 15) * ld + ((lane >> 4) << 3);
  union { u32x4 u[2]; v16bf v; } x;
  x.u[0] = *(const u32x4*)(p);
  x.u[1] = *(const u32x4*)(p + 16);
  return x.v;
}
__device__ __forceinline__ v8f wmma_bf(v16bf a, v16bf b, v8f c) {
  return __builtin_amdgcn_wmma_f32_16x16x32_bf16(false, a, false, b, (short)0, c,
                                                 false, false);
}

// ---- async global->LDS copy (16B per lane, tracked by ASYNCcnt) ------------
// LDS operand is the low 32 bits of the flat shared address.
__device__ __forceinline__ void async_ld_b128(void* lds_dst, const void* gsrc) {
  unsigned lds_off = (unsigned)(uintptr_t)lds_dst;
  asm volatile("global_load_async_to_lds_b128 %0, %1, off"
               :: "v"(lds_off), "v"(gsrc) : "memory");
}
__device__ __forceinline__ void wait_async0() {
  asm volatile("s_wait_asynccnt 0x0" ::: "memory");
}

// ---- elementwise converts --------------------------------------------------
__global__ __launch_bounds__(256) void k_cvt(const float* __restrict__ x,
                                             bf16_t* __restrict__ o,
                                             float scale, size_t n) {
  size_t i = (size_t)blockIdx.x * 256 + threadIdx.x;
  if (i < n) o[i] = (bf16_t)(x[i] * scale);
}
__global__ __launch_bounds__(256) void k_prod(const float* __restrict__ a,
                                              const float* __restrict__ b,
                                              bf16_t* __restrict__ o, size_t n) {
  size_t i = (size_t)blockIdx.x * 256 + threadIdx.x;
  if (i < n) o[i] = (bf16_t)(a[i] * b[i]);
}

// ---- generic GEMM: C[MxN] = A[MxK](bf16) * W^T + bias, W is NxK row-major --
// block = 128 threads (4 waves); block tile 64 rows x 64 cols; wave = 16x64.
// WB/WF compile-time: write bf16 / write f32 outputs (no epilogue branches).
template <bool WB, bool WF>
__global__ __launch_bounds__(128) void k_gemm(const bf16_t* __restrict__ A,
                                              const bf16_t* __restrict__ W,
                                              const float* __restrict__ bias,
                                              bf16_t* __restrict__ Cb,
                                              float* __restrict__ Cf,
                                              int N, int K) {
  const int lane = threadIdx.x & 31;
  const int wave = threadIdx.x >> 5;
  const int row0 = (blockIdx.x * 4 + wave) * 16;
  const int col0 = blockIdx.y * 64;
  v16bf a[8];
#pragma unroll
  for (int ks = 0; ks < 8; ++ks)
    a[ks] = frag_a(A + (size_t)row0 * K + ks * 32, K, lane);
#pragma unroll
  for (int ct = 0; ct < 4; ++ct) {
    const int cc = col0 + ct * 16;
    v8f acc = {};
#pragma unroll
    for (int ks = 0; ks < 8; ++ks) {
      v16bf b = frag_b(W + (size_t)cc * K + ks * 32, K, lane);
      acc = wmma_bf(a[ks], b, acc);
    }
    const int col  = cc + (lane & 15);
    const float bv = bias[col];
    const int rb   = row0 + ((lane >> 4) << 3);
#pragma unroll
    for (int i = 0; i < 8; ++i) {
      float v = acc[i] + bv;
      size_t idx = (size_t)(rb + i) * N + col;
      if (WF) Cf[idx] = v;
      if (WB) Cb[idx] = (bf16_t)v;
    }
  }
}

// ---- batch-dim attention: one block per sequence position n ----------------
// scores[l,m] = q[l,n,:].k[m,n,:]/16 (32x32, K=256); softmax rows; O = attn.V
__global__ __launch_bounds__(128) void k_attn(const bf16_t* __restrict__ qkv_tf,
                                              const bf16_t* __restrict__ qkv_gf,
                                              bf16_t* __restrict__ o_tf,
                                              bf16_t* __restrict__ o_gf) {
  const int  n      = blockIdx.x;
  const bool second = blockIdx.y != 0;
  const bf16_t* Qm = second ? qkv_gf : qkv_tf;  // q from self
  const bf16_t* Km = second ? qkv_tf : qkv_gf;  // k from other
  const bf16_t* Vm = second ? qkv_gf : qkv_tf;  // v from self
  bf16_t*       Om = second ? o_gf   : o_tf;

  __shared__ __align__(16) bf16_t Qs[32][256];
  __shared__ __align__(16) bf16_t Ks[32][256];
  __shared__ __align__(16) bf16_t Vt[256][32];  // transposed V: [e][m]
  __shared__ __align__(16) float  Sc[32][32];
  __shared__ __align__(16) bf16_t At[32][32];

  const int tid = threadIdx.x;
  for (int c = tid; c < 1024; c += 128) {
    const int row = c >> 5;         // batch index l/m 0..31
    const int off = (c & 31) * 8;   // embedding chunk
    const size_t g = ((size_t)row * SEQ + n) * 768;
    // Q, K: async DMA straight into LDS (no VGPR round-trip)
    async_ld_b128(&Qs[row][off], Qm + g + 0   + off);
    async_ld_b128(&Ks[row][off], Km + g + 256 + off);
    // V: register transpose into [e][m] layout for the B-fragment
    u32x4 v = *(const u32x4*)(Vm + g + 512 + off);
    const bf16_t* pv = (const bf16_t*)&v;
#pragma unroll
    for (int j = 0; j < 8; ++j) Vt[off + j][row] = pv[j];
  }
  wait_async0();
  __syncthreads();

  {  // scores: 4 waves, each one 16x16 tile of the 32x32 score matrix
    const int wv = tid >> 5, lane = tid & 31;
    const int ri = wv >> 1, ci = wv & 1;
    v8f acc = {};
#pragma unroll
    for (int ks = 0; ks < 8; ++ks) {
      v16bf a = frag_a(&Qs[ri * 16][ks * 32], 256, lane);
      v16bf b = frag_b(&Ks[ci * 16][ks * 32], 256, lane);  // K rows are B^T
      acc = wmma_bf(a, b, acc);
    }
    const int col = ci * 16 + (lane & 15);
    const int rb  = ri * 16 + ((lane >> 4) << 3);
#pragma unroll
    for (int i = 0; i < 8; ++i) Sc[rb + i][col] = acc[i] * 0.0625f;  // 1/sqrt(256)
  }
  __syncthreads();

  if (tid < 32) {  // row softmax
    float m = -1e30f;
#pragma unroll
    for (int j = 0; j < 32; ++j) m = fmaxf(m, Sc[tid][j]);
    float e[32], s = 0.f;
#pragma unroll
    for (int j = 0; j < 32; ++j) { e[j] = __expf(Sc[tid][j] - m); s += e[j]; }
    const float inv = 1.f / s;
#pragma unroll
    for (int j = 0; j < 32; ++j) At[tid][j] = (bf16_t)(e[j] * inv);
  }
  __syncthreads();

  {  // O = attn (32x32) @ V (32x256): 32 tiles, 8 per wave, one WMMA each
    const int wv = tid >> 5, lane = tid & 31;
    const int rh = wv & 1;
    v16bf a = frag_a(&At[rh * 16][0], 32, lane);
#pragma unroll
    for (int j = 0; j < 8; ++j) {
      const int ct = (wv >> 1) * 8 + j;  // 0..15
      v16bf b = frag_b(&Vt[ct * 16][0], 32, lane);
      v8f acc = {};
      acc = wmma_bf(a, b, acc);
      const int col = ct * 16 + (lane & 15);
      const int rb  = rh * 16 + ((lane >> 4) << 3);
#pragma unroll
      for (int i = 0; i < 8; ++i)
        Om[((size_t)(rb + i) * SEQ + n) * 256 + col] = (bf16_t)acc[i];
    }
  }
}

// ---- fused gh-GEMM + GRU gates --------------------------------------------
// gh[:,j|j+256|j+512] = h @ Whh^T + bhh ; gates with precomputed gi (bias in)
__global__ __launch_bounds__(128) void k_gru(const float* __restrict__ gi,
                                             const bf16_t* __restrict__ hb,
                                             const float* __restrict__ hf,
                                             const bf16_t* __restrict__ Whh,
                                             const float* __restrict__ bhh,
                                             float* __restrict__ g) {
  const int lane = threadIdx.x & 31;
  const int wave = threadIdx.x >> 5;
  const int row0 = (blockIdx.x * 4 + wave) * 16;
  const int col0 = blockIdx.y * 16;
  v16bf a[8];
#pragma unroll
  for (int ks = 0; ks < 8; ++ks)
    a[ks] = frag_a(hb + (size_t)row0 * 256 + ks * 32, 256, lane);
  v8f a0 = {}, a1 = {}, a2 = {};
#pragma unroll
  for (int ks = 0; ks < 8; ++ks) {
    v16bf b0 = frag_b(Whh + (size_t)(col0 + 0)   * 256 + ks * 32, 256, lane);
    v16bf b1 = frag_b(Whh + (size_t)(col0 + 256) * 256 + ks * 32, 256, lane);
    v16bf b2 = frag_b(Whh + (size_t)(col0 + 512) * 256 + ks * 32, 256, lane);
    a0 = wmma_bf(a[ks], b0, a0);
    a1 = wmma_bf(a[ks], b1, a1);
    a2 = wmma_bf(a[ks], b2, a2);
  }
  const int col = col0 + (lane & 15);
  const int rb  = row0 + ((lane >> 4) << 3);
  const float b0 = bhh[col], b1 = bhh[col + 256], b2 = bhh[col + 512];
#pragma unroll
  for (int i = 0; i < 8; ++i) {
    const size_t m = (size_t)(rb + i);
    const float ir  = gi[m * 768 + col];
    const float iz  = gi[m * 768 + col + 256];
    const float in_ = gi[m * 768 + col + 512];
    const float r = 1.f / (1.f + __expf(-(ir + (a0[i] + b0))));
    const float z = 1.f / (1.f + __expf(-(iz + (a1[i] + b1))));
    const float nn = tanhf(in_ + r * (a2[i] + b2));
    const float hv = hf[m * 256 + col];
    g[m * 256 + col] = (1.f - z) * nn + z * hv;
  }
}

// ---- fused dual 7x7 conv + relu + sum, LDS tiled ---------------------------
__global__ __launch_bounds__(256) void k_conv(const float* __restrict__ g1,
                                              const float* __restrict__ g2,
                                              const float* __restrict__ w49,
                                              const float* __restrict__ bptr,
                                              float* __restrict__ out) {
  __shared__ float t1[14][40];
  __shared__ float t2[14][40];
  __shared__ float ws[49];
  const int b  = blockIdx.z;
  const int s0 = blockIdx.y * 8;
  const int e0 = blockIdx.x * 32;
  const int tid = threadIdx.y * 32 + threadIdx.x;
  if (tid < 49) ws[tid] = w49[tid];
  const size_t base = (size_t)b * SEQ * EMB;
  for (int idx = tid; idx < 14 * 38; idx += 256) {
    const int ls = idx / 38, le = idx % 38;
    const int gs = s0 - 3 + ls, ge = e0 - 3 + le;
    const bool ok = (gs >= 0 && gs < SEQ && ge >= 0 && ge < EMB);
    const size_t gi = base + (size_t)gs * EMB + ge;
    t1[ls][le] = ok ? g1[gi] : 0.f;
    t2[ls][le] = ok ? g2[gi] : 0.f;
  }
  __syncthreads();
  const float sb = bptr[0];
  const int ls = threadIdx.y, le = threadIdx.x;
  float s1 = 0.f, s2 = 0.f;
#pragma unroll
  for (int dy = 0; dy < 7; ++dy)
#pragma unroll
    for (int dx = 0; dx < 7; ++dx) {
      const float wv = ws[dy * 7 + dx];
      s1 += wv * t1[ls + dy][le + dx];
      s2 += wv * t2[ls + dy][le + dx];
    }
  out[base + (size_t)(s0 + ls) * EMB + (e0 + le)] =
      fmaxf(s1 + sb, 0.f) + fmaxf(s2 + sb, 0.f);
}

// ---------------------------------------------------------------------------
extern "C" void kernel_launch(void* const* d_in, const int* in_sizes, int n_in,
                              void* d_out, int out_size, void* d_ws, size_t ws_size,
                              hipStream_t stream) {
  const float* text  = (const float*)d_in[0];
  const float* graph = (const float*)d_in[1];
  // d_in[2..7] (avg/max/sp params) are mathematically dead — see header note.
  const float* in_w  = (const float*)d_in[8];
  const float* in_b  = (const float*)d_in[9];
  const float* out_w = (const float*)d_in[10];
  const float* out_b = (const float*)d_in[11];
  const float* lin_w = (const float*)d_in[12];
  const float* lin_b = (const float*)d_in[13];
  const float* w_ih  = (const float*)d_in[14];
  const float* w_hh  = (const float*)d_in[15];
  const float* b_ih  = (const float*)d_in[16];
  const float* b_hh  = (const float*)d_in[17];
  const float* ssf_w = (const float*)d_in[18];
  const float* ssf_b = (const float*)d_in[19];
  float* outp = (float*)d_out;

  const size_t M  = MROWS;
  const size_t NB = M * 256;         // elements per (M x 256) matrix

  // Workspace layout with lifetime aliasing (peak ~578 MiB).
  char* ws = (char*)d_ws;
  size_t off = 0;
  auto take = [&](size_t bytes) { char* p = ws + off; off += (bytes + 255) & ~(size_t)255; return p; };
  bf16_t* tfb      = (bf16_t*)take(NB * 2);          // region A (reused: g_gf)
  bf16_t* gfb      = (bf16_t*)take(NB * 2);
  bf16_t* qkv_tf   = (bf16_t*)take(M * 768 * 2);     // region B (reused: gi)
  bf16_t* qkv_gf   = (bf16_t*)take(M * 768 * 2);
  bf16_t* o_tf     = (bf16_t*)take(NB * 2);          // region C (reused: g_tf)
  bf16_t* o_gf     = (bf16_t*)take(NB * 2);
  float*  att_tf_f = (float*)take(NB * 4);
  float*  att_gf_f = (float*)take(NB * 4);
  bf16_t* att_tf_b = (bf16_t*)take(NB * 2);
  bf16_t* att_gf_b = (bf16_t*)take(NB * 2);
  bf16_t* prod_b   = (bf16_t*)take(NB * 2);
  bf16_t* inter_b  = (bf16_t*)take(NB * 2);
  bf16_t* Wqkv     = (bf16_t*)take(768 * 256 * 2);
  bf16_t* Wout     = (bf16_t*)take(256 * 256 * 2);
  bf16_t* Wlin     = (bf16_t*)take(256 * 256 * 2);
  bf16_t* Wih      = (bf16_t*)take(768 * 256 * 2);
  bf16_t* Whh      = (bf16_t*)take(768 * 256 * 2);
  // aliases (inputs of these stages are disjoint from the aliased regions)
  float* gi   = (float*)qkv_tf;   // M x 768 fp32 = 192 MiB over qkv_tf+qkv_gf
  float* g_tf = (float*)o_tf;     // M x 256 fp32 = 64 MiB over o_tf+o_gf
  float* g_gf = (float*)tfb;      // M x 256 fp32 = 64 MiB over tfb+gfb
  (void)ws_size; (void)out_size; (void)n_in; (void)in_sizes;

  // 1) weight converts (bf16, kept N x K row-major == ready-to-use B^T)
  k_cvt<<<dim3((768 * 256 + 255) / 256), 256, 0, stream>>>(in_w,  Wqkv, 1.f, 768 * 256);
  k_cvt<<<dim3((256 * 256 + 255) / 256), 256, 0, stream>>>(out_w, Wout, 1.f, 256 * 256);
  k_cvt<<<dim3((256 * 256 + 255) / 256), 256, 0, stream>>>(lin_w, Wlin, 1.f, 256 * 256);
  k_cvt<<<dim3((768 * 256 + 255) / 256), 256, 0, stream>>>(w_ih,  Wih,  1.f, 768 * 256);
  k_cvt<<<dim3((768 * 256 + 255) / 256), 256, 0, stream>>>(w_hh,  Whh,  1.f, 768 * 256);
  // 2) fused front-end: tf_fused = 2*text, gf_fused = 2*graph (exact)
  k_cvt<<<dim3(NB / 256), 256, 0, stream>>>(text,  tfb, 2.f, NB);
  k_cvt<<<dim3(NB / 256), 256, 0, stream>>>(graph, gfb, 2.f, NB);
  // 3) QKV projections (N = 768)
  k_gemm<true, false><<<dim3(M / 64, 12), 128, 0, stream>>>(tfb, Wqkv, in_b, qkv_tf, nullptr, 768, 256);
  k_gemm<true, false><<<dim3(M / 64, 12), 128, 0, stream>>>(gfb, Wqkv, in_b, qkv_gf, nullptr, 768, 256);
  // 4) batch-dim attention (y=0: tf_att path, y=1: gf_att path)
  k_attn<<<dim3(SEQ, 2), 128, 0, stream>>>(qkv_tf, qkv_gf, o_tf, o_gf);
  // 5) output projections -> fp32 (for elemwise/GRU h) + bf16 (for gh GEMM)
  k_gemm<true, true><<<dim3(M / 64, 4), 128, 0, stream>>>(o_tf, Wout, out_b, att_tf_b, att_tf_f, 256, 256);
  k_gemm<true, true><<<dim3(M / 64, 4), 128, 0, stream>>>(o_gf, Wout, out_b, att_gf_b, att_gf_f, 256, 256);
  // 6) inter = (tf_att * gf_att) @ lin_w^T + lin_b
  k_prod<<<dim3(NB / 256), 256, 0, stream>>>(att_tf_f, att_gf_f, prod_b, NB);
  k_gemm<true, false><<<dim3(M / 64, 4), 128, 0, stream>>>(prod_b, Wlin, lin_b, inter_b, nullptr, 256, 256);
  // 7) gi = inter @ w_ih^T + b_ih (shared by both GRU cells)
  k_gemm<false, true><<<dim3(M / 64, 12), 128, 0, stream>>>(inter_b, Wih, b_ih, nullptr, gi, 768, 256);
  // 8) fused gh GEMM + GRU gates
  k_gru<<<dim3(M / 64, 16), 128, 0, stream>>>(gi, att_tf_b, att_tf_f, Whh, b_hh, g_tf);
  k_gru<<<dim3(M / 64, 16), 128, 0, stream>>>(gi, att_gf_b, att_gf_f, Whh, b_hh, g_gf);
  // 9) out = relu(conv7x7(g_tf)+b) + relu(conv7x7(g_gf)+b)
  k_conv<<<dim3(EMB / 32, SEQ / 8, BAT), dim3(32, 8), 0, stream>>>(g_tf, g_gf, ssf_w, ssf_b, outp);
}
